// APLoss_83528523973056
// MI455X (gfx1250) — matrix operand order repaired and checked by reading, not compiled, not run on hardware
//
#include <hip/hip_runtime.h>
#include <hip/hip_bf16.h>

typedef _Float16 v16h __attribute__((ext_vector_type(16)));
typedef float    v8f  __attribute__((ext_vector_type(8)));

#define BATCH 2
#define CCH   128
#define HH    48
#define WW    48
#define NPIX  (HH*WW)        // 2304
#define NQB   25
#define RAD   4
#define KREL  0.5f

#define NB     64            // query columns per workgroup (4 waves x 16)
#define MCHUNK 144           // db rows per chunk (one LDS stage); 144 = 3*48
#define MTILES (MCHUNK/16)   // 9

union V16U { v16h v; uint4 q[2]; };

// ---------------------------------------------------------------------------
// Pass 0: zero bin partial arrays + output scalar
// ---------------------------------------------------------------------------
__global__ void init_kernel(float* __restrict__ Pn, float* __restrict__ Pr,
                            float* __restrict__ out) {
    int i = blockIdx.x * blockDim.x + threadIdx.x;
    int total = BATCH * NPIX * NQB;
    if (i < total) { Pn[i] = 0.f; Pr[i] = 0.f; }
    if (i == 0) out[0] = 0.f;
}

// ---------------------------------------------------------------------------
// Pass 1: bilinear grid-sample of image2 -> Dh (fp16, row-major N x C),
//         transpose-convert image1 -> Qh (fp16, row-major N x C)
// ---------------------------------------------------------------------------
__global__ __launch_bounds__(CCH)
void prep_kernel(const float* __restrict__ img1, const float* __restrict__ img2,
                 const float* __restrict__ grid,
                 _Float16* __restrict__ Qh, _Float16* __restrict__ Dh) {
    const int c  = threadIdx.x;
    const int bj = blockIdx.x;
    const int b  = bj / NPIX;
    const int j  = bj % NPIX;

    Qh[((size_t)b * NPIX + j) * CCH + c] =
        (_Float16)img1[((size_t)(b * CCH + c)) * NPIX + j];

    float gx = grid[(size_t)bj * 2 + 0];
    float gy = grid[(size_t)bj * 2 + 1];
    float ix = ((gx + 1.f) * (float)WW - 1.f) * 0.5f;
    float iy = ((gy + 1.f) * (float)HH - 1.f) * 0.5f;
    float x0f = floorf(ix), y0f = floorf(iy);
    float wx1 = ix - x0f,   wy1 = iy - y0f;
    int x0 = (int)x0f, y0 = (int)y0f;
    const float* base = img2 + ((size_t)(b * CCH + c)) * NPIX;
    float val = 0.f;
#pragma unroll
    for (int dy = 0; dy < 2; ++dy) {
#pragma unroll
        for (int dx = 0; dx < 2; ++dx) {
            int xi = x0 + dx, yi = y0 + dy;
            bool valid = (xi >= 0) && (xi < WW) && (yi >= 0) && (yi < HH);
            float w = (dx ? wx1 : 1.f - wx1) * (dy ? wy1 : 1.f - wy1);
            int xc = min(max(xi, 0), WW - 1);
            int yc = min(max(yi, 0), HH - 1);
            val = fmaf(valid ? w : 0.f, base[yc * WW + xc], val);
        }
    }
    Dh[((size_t)b * NPIX + j) * CCH + c] = (_Float16)val;
}

// ---------------------------------------------------------------------------
// Pass 2: S^T tiles via v_wmma_f32_16x16x32_f16 + fused quantized-AP bins.
// LDS staging of the D chunk uses CDNA5 GLOBAL_LOAD_ASYNC_TO_LDS_B128
// (ASYNCcnt-tracked DMA into LDS, no VGPR round trip).
// grid = (NPIX/NB, NPIX/MCHUNK, BATCH), block = 128 (4 waves)
// ---------------------------------------------------------------------------
__global__ __launch_bounds__(128)
void gemm_ap_kernel(const _Float16* __restrict__ Qh,
                    const _Float16* __restrict__ Dh,
                    float* __restrict__ Pn, float* __restrict__ Pr) {
    __shared__ _Float16 Ds[MCHUNK * CCH];   // 36 KB of the 320 KB/WGP LDS

    const int tid  = threadIdx.x;
    const int lane = tid & 31;
    const int b    = blockIdx.z;
    const int m0c  = blockIdx.y * MCHUNK;
    const int n0w  = blockIdx.x * NB + (tid >> 5) * 16;

    // --- async DMA stage: global -> LDS (144 rows x 128 fp16 = 36864 B) ---
    {
        const char* srcbase = (const char*)(Dh + ((size_t)b * NPIX + m0c) * CCH);
        unsigned ldsbase = (unsigned)(size_t)(&Ds[0]);  // low 32 bits = LDS offset
        for (int i = tid; i < MCHUNK * CCH / 8; i += 128) {
            unsigned voff  = (unsigned)i * 16u;   // byte offset into chunk
            unsigned laddr = ldsbase + voff;
            asm volatile("global_load_async_to_lds_b128 %0, %1, %2 offset:0"
                         :: "v"(laddr), "v"(voff), "s"(srcbase) : "memory");
        }
        asm volatile("s_wait_asynccnt 0x0" ::: "memory");
    }
    __syncthreads();

    const int halfsel = lane >> 4;   // 0 or 1 (wave half)
    const int lcol    = lane & 15;
    const int n  = n0w + lcol;       // this lane's query row
    const int rn = n / WW, cn = n % WW;

    // B fragments (Q rows as B columns), register-resident across the m loop.
    const _Float16* qrow = Qh + ((size_t)b * NPIX + n) * CCH;
    V16U Bf[4];
#pragma unroll
    for (int kc = 0; kc < 4; ++kc) {
        const _Float16* p = qrow + kc * 32 + halfsel * 8;
        Bf[kc].q[0] = *(const uint4*)(p);
        Bf[kc].q[1] = *(const uint4*)(p + 16);
    }

    // cumulative-bin accumulators: CQ_c(s) = clamp(12 s + c - 11, 0, 1)
    float accN[NQB], accR[NQB];
#pragma unroll
    for (int c = 0; c < NQB; ++c) { accN[c] = 0.f; accR[c] = 0.f; }

    // incremental (row, col) of the 8-run base: m0c divisible by 48, and all
    // 8-runs are 8-aligned with 48 % 8 == 0, so a run never crosses a row.
    int rm0 = blockIdx.y * 3;        // m0c / 48  (MCHUNK = 3*48)
    int cm0 = halfsel * 8;           // m0c % 48 + halfsel*8

    for (int mt = 0; mt < MTILES; ++mt) {
        const _Float16* arow = Ds + (mt * 16 + lcol) * CCH;
        v8f acc0 = {}, acc1 = {};
#pragma unroll
        for (int kp = 0; kp < 2; ++kp) {   // two independent 2-deep chains
            V16U Af;
            const _Float16* p0 = arow + (kp * 2) * 32 + halfsel * 8;
            Af.q[0] = *(const uint4*)(p0);
            Af.q[1] = *(const uint4*)(p0 + 16);
            acc0 = __builtin_amdgcn_wmma_f32_16x16x32_f16(
                false, Af.v, false, Bf[kp * 2].v, (short)0, acc0, false, false);
            const _Float16* p1 = arow + (kp * 2 + 1) * 32 + halfsel * 8;
            Af.q[0] = *(const uint4*)(p1);
            Af.q[1] = *(const uint4*)(p1 + 16);
            acc1 = __builtin_amdgcn_wmma_f32_16x16x32_f16(
                false, Af.v, false, Bf[kp * 2 + 1].v, (short)0, acc1, false, false);
        }
        v8f acc = acc0 + acc1;

        // labels for this 8-run (division-free)
        float labs[8];
        int anylab = 0;
#pragma unroll
        for (int r = 0; r < 8; ++r) {
            int du = rm0 - rn + RAD;        // row of run is constant
            int dv = cm0 + r - cn + RAD;
            float lab = ((unsigned)du <= 2u * RAD && (unsigned)dv <= 2u * RAD)
                        ? 1.f : 0.f;
            labs[r] = lab;
            anylab |= (lab != 0.f);
        }

        if (__any(anylab)) {                // ~15% of tiles
#pragma unroll
            for (int r = 0; r < 8; ++r) {
                float basev = fmaf(12.f, acc[r], -11.f);
                float lab = labs[r];
#pragma unroll
                for (int c = 0; c < NQB; ++c) {
                    float qc = fminf(fmaxf(basev + (float)c, 0.f), 1.f);
                    accN[c] += qc;
                    accR[c]  = fmaf(lab, qc, accR[c]);
                }
            }
        } else {                            // labels all zero: nbs only
#pragma unroll
            for (int r = 0; r < 8; ++r) {
                float basev = fmaf(12.f, acc[r], -11.f);
#pragma unroll
                for (int c = 0; c < NQB; ++c) {
                    float qc = fminf(fmaxf(basev + (float)c, 0.f), 1.f);
                    accN[c] += qc;
                }
            }
        }

        cm0 += 16;                          // advance run base by one m-tile
        if (cm0 >= WW) { cm0 -= WW; rm0 += 1; }
    }

    // combine the two wave halves (same query row in lanes L and L+16)
#pragma unroll
    for (int c = 0; c < NQB; ++c) {
        accN[c] += __shfl_xor(accN[c], 16, 32);
        accR[c] += __shfl_xor(accR[c], 16, 32);
    }
    if (lane < 16) {
        float* pn = Pn + ((size_t)b * NPIX + n) * NQB;
        float* pr = Pr + ((size_t)b * NPIX + n) * NQB;
#pragma unroll
        for (int c = 0; c < NQB; ++c) {
            atomicAdd(&pn[c], accN[c]);
            atomicAdd(&pr[c], accR[c]);
        }
    }
}

// ---------------------------------------------------------------------------
// Pass 3: per-row AP from cumulative bins, reliability weighting, scalar mean
// ---------------------------------------------------------------------------
__global__ void finalize_kernel(const float* __restrict__ Pn,
                                const float* __restrict__ Pr,
                                const float* __restrict__ rel,
                                float* __restrict__ out) {
    int i = blockIdx.x * blockDim.x + threadIdx.x;
    if (i >= BATCH * NPIX) return;
    const float* pn = Pn + (size_t)i * NQB;
    const float* pr = Pr + (size_t)i * NQB;
    float prevR = 0.f, apnum = 0.f;
#pragma unroll
    for (int c = 0; c < NQB; ++c) {
        float cumN = pn[c];          // cumsum(nbs)[c]
        float cumR = pr[c];          // cumsum(rec)[c]
        float recc = cumR - prevR;   // rec[c]
        float prec = cumR / (1e-16f + cumN);
        apnum = fmaf(prec, recc, apnum);
        prevR = cumR;
    }
    float ap = apnum / prevR;        // prevR == sum(rec) >= 25, never 0
    float r  = rel[i];
    float loss = 1.f - (ap * r + KREL * (1.f - r));
    atomicAdd(out, loss * (1.f / (float)(BATCH * NPIX)));
}

// ---------------------------------------------------------------------------
extern "C" void kernel_launch(void* const* d_in, const int* in_sizes, int n_in,
                              void* d_out, int out_size, void* d_ws, size_t ws_size,
                              hipStream_t stream) {
    (void)in_sizes; (void)n_in; (void)out_size; (void)ws_size;
    const float* img1 = (const float*)d_in[0];  // (B,C,H,W)
    const float* img2 = (const float*)d_in[1];  // (B,C,H,W)
    const float* rel  = (const float*)d_in[2];  // (B,1,H,W)
    const float* grid = (const float*)d_in[3];  // (B,H,W,2)

    char* ws = (char*)d_ws;
    const size_t szQ = (size_t)BATCH * NPIX * CCH * 2;            // fp16 bytes
    const size_t szP = (size_t)BATCH * NPIX * NQB * sizeof(float);
    _Float16* Qh = (_Float16*)(ws);
    _Float16* Dh = (_Float16*)(ws + szQ);
    float*    Pn = (float*)(ws + 2 * szQ);
    float*    Pr = (float*)(ws + 2 * szQ + szP);
    float*    out = (float*)d_out;

    int total = BATCH * NPIX * NQB;
    init_kernel<<<(total + 255) / 256, 256, 0, stream>>>(Pn, Pr, out);
    prep_kernel<<<BATCH * NPIX, CCH, 0, stream>>>(img1, img2, grid, Qh, Dh);
    dim3 g(NPIX / NB, NPIX / MCHUNK, BATCH);
    gemm_ap_kernel<<<g, 128, 0, stream>>>(Qh, Dh, Pn, Pr);
    finalize_kernel<<<(BATCH * NPIX + 255) / 256, 256, 0, stream>>>(Pn, Pr, rel, out);
}
